// FastAttention_82824149336914
// MI455X (gfx1250) — compile-verified, matrix operand order checked by reading, not run on hardware
//
#include <hip/hip_runtime.h>

// ---------- types ----------
typedef _Float16 f16;
typedef __attribute__((ext_vector_type(16))) _Float16 v16h;
typedef __attribute__((ext_vector_type(8)))  _Float16 v8h;
typedef __attribute__((ext_vector_type(4)))  _Float16 h4;
typedef __attribute__((ext_vector_type(8)))  float    v8f;
typedef __attribute__((ext_vector_type(4)))  float    f32x4;

// ---------- problem constants ----------
constexpr int Bc = 4, Hc = 16, Lc = 4096, Dc = 64, Mc = 256;
constexpr int BH = Bc * Hc;          // 64 batched heads
constexpr int EP = 80;               // 64 V cols + 1 den col, padded to 5 N-tiles
constexpr int SC = 256;              // K/V rows per workgroup (kernel A)
constexpr int QC = 128;              // Q rows per workgroup (kernel B)
constexpr int DP = Dc + 8;           // LDS row pad (halves): 36 dwords -> conflict-free b128
constexpr int CP = Mc + 8;           // LDS row pad for cT/buf1T: 132 dwords
constexpr float NM   = 0.35355339059327373f;   // 64^-0.25
constexpr float LN16 = 2.772588722239781f;     // ln(sqrt(M)) with M=256
constexpr float EPSV = 1e-6f;

// ---------- WMMA helpers ----------
__device__ __forceinline__ v8f wmma(v16h a, v16h b, v8f c) {
  return __builtin_amdgcn_wmma_f32_16x16x32_f16(false, a, false, b, (short)0, c,
                                                false, false);
}

// Load one 16-lane-striped f16 fragment (A row-major / B column-major pattern):
// halves 0..7  <- lds[row*stride + k0 + 8*hi + j]
// halves 8..15 <- lds[row*stride + k0 + 16 + 8*hi + j]
__device__ __forceinline__ v16h ldsfrag(const f16* p, int row, int stride,
                                        int k0, int hi) {
  int base = row * stride + k0 + (hi << 3);
  v16h r;
  ((v8h*)&r)[0] = *(const v8h*)(p + base);
  ((v8h*)&r)[1] = *(const v8h*)(p + base + 16);
  return r;
}

// =====================================================================
// Kernel A: buf1[head][m][e] += sum_s ksp[s,m] * c[s,e]   (c = [V*mask | mask])
// grid = BH * (L/SC) = 1024 blocks of 256 threads (8 waves)
// =====================================================================
__global__ __launch_bounds__(256, 1)
void perf_bufk(const float* __restrict__ ks, const float* __restrict__ vs,
               const float* __restrict__ mask, const float* __restrict__ proj,
               float* __restrict__ buf1) {
  __shared__ __align__(16) f16  kt[SC * DP];   // scaled K chunk, row-major
  __shared__ __align__(16) f16  cT[EP * CP];   // c transposed: [e][s]
  __shared__ __align__(16) f16  pj[Mc * DP];   // proj, row-major [m][d]
  __shared__ float hn[SC];                     // 0.5|x|^2 + ln16 per row

  const int tid   = threadIdx.x;
  const int head  = blockIdx.x >> 4;           // L/SC = 16 chunks
  const int chunk = blockIdx.x & 15;
  const int sbase = chunk * SC;
  const int bidx  = head >> 4;                 // head / H

  // --- stage K row (scale, sum-of-squares, f16) ---
  {
    const int s = tid;
    const float* kr = ks + ((size_t)head * Lc + sbase + s) * Dc;
    f16* dst = kt + s * DP;
    float ss = 0.f;
#pragma unroll
    for (int i = 0; i < 16; ++i) {
      f32x4 x = ((const f32x4*)kr)[i];
      x.x *= NM; x.y *= NM; x.z *= NM; x.w *= NM;
      ss += x.x * x.x + x.y * x.y + x.z * x.z + x.w * x.w;
      h4 y = {(f16)x.x, (f16)x.y, (f16)x.z, (f16)x.w};
      ((h4*)dst)[i] = y;
    }
    hn[s] = 0.5f * ss + LN16;
  }
  // --- stage V*mask transposed + ones column ---
  {
    const int s = tid;
    const float* vr = vs + ((size_t)head * Lc + sbase + s) * Dc;
    const float mv = mask[(size_t)bidx * Lc + sbase + s];
#pragma unroll 8
    for (int e = 0; e < Dc; ++e) cT[e * CP + s] = (f16)(vr[e] * mv);
    cT[64 * CP + s] = (f16)mv;
#pragma unroll
    for (int e = 65; e < EP; ++e) cT[e * CP + s] = (f16)0.f;
  }
  // --- stage proj ---
  {
    const int m = tid;
    const float* pr = proj + m * Dc;
    f16* dst = pj + m * DP;
#pragma unroll
    for (int i = 0; i < 16; ++i) {
      f32x4 x = ((const f32x4*)pr)[i];
      h4 y = {(f16)x.x, (f16)x.y, (f16)x.z, (f16)x.w};
      ((h4*)dst)[i] = y;
    }
  }
  __syncthreads();

  const int lane = tid & 31, w = tid >> 5;
  const int l15 = lane & 15, hi = lane >> 4;

  // resident proj B-fragments for this wave's two m-tiles
  v16h pb[2][2];
#pragma unroll
  for (int mi = 0; mi < 2; ++mi)
#pragma unroll
    for (int kk = 0; kk < 2; ++kk)
      pb[mi][kk] = ldsfrag(pj, (2 * w + mi) * 16 + l15, DP, kk * 32, hi);

  v8f d2[2][5];
#pragma unroll
  for (int mi = 0; mi < 2; ++mi)
#pragma unroll
    for (int et = 0; et < 5; ++et) { v8f z = {}; d2[mi][et] = z; }

  for (int it = 0; it < SC / 32; ++it) {
    const int s0 = it * 32;
    float hnv[2][8];
#pragma unroll
    for (int sti = 0; sti < 2; ++sti)
#pragma unroll
      for (int v = 0; v < 8; ++v)
        hnv[sti][v] = hn[s0 + sti * 16 + hi * 8 + v];

    // WMMA1: xp tiles [s x m], contraction over d (2 x k32)
    v8f d1[2][2];
#pragma unroll
    for (int sti = 0; sti < 2; ++sti) {
      v16h a0 = ldsfrag(kt, s0 + sti * 16 + l15, DP, 0, hi);
      v16h a1 = ldsfrag(kt, s0 + sti * 16 + l15, DP, 32, hi);
#pragma unroll
      for (int mi = 0; mi < 2; ++mi) {
        v8f z = {};
        z = wmma(a0, pb[mi][0], z);
        z = wmma(a1, pb[mi][1], z);
        d1[sti][mi] = z;
      }
    }
    // phi in registers -> A2 fragments (rows = m, halves = s)
    v16h a2[2];
#pragma unroll
    for (int mi = 0; mi < 2; ++mi)
#pragma unroll
      for (int v = 0; v < 8; ++v) {
        a2[mi][v]     = (f16)__expf(d1[0][mi][v] - hnv[0][v]);
        a2[mi][8 + v] = (f16)__expf(d1[1][mi][v] - hnv[1][v]);
      }
    // WMMA2: accumulate buf1 tiles over s
#pragma unroll
    for (int et = 0; et < 5; ++et) {
      v16h b2 = ldsfrag(cT, et * 16 + l15, CP, s0, hi);
      d2[0][et] = wmma(a2[0], b2, d2[0][et]);
      d2[1][et] = wmma(a2[1], b2, d2[1][et]);
    }
  }

  // accumulate partial buf1 (only e < 65 is meaningful)
  float* bh = buf1 + (size_t)head * Mc * EP;
#pragma unroll
  for (int mi = 0; mi < 2; ++mi)
#pragma unroll
    for (int et = 0; et < 5; ++et) {
      const int e = et * 16 + l15;
      if (e < 65) {
#pragma unroll
        for (int v = 0; v < 8; ++v) {
          const int m = (2 * w + mi) * 16 + hi * 8 + v;
          unsafeAtomicAdd(bh + m * EP + e, d2[mi][et][v]);
        }
      }
    }
}

// =====================================================================
// Kernel B: ctx[q, :] = (qsp[q,:] @ buf1)[:64] / max(den, eps)
// grid = BH * (L/QC) = 2048 blocks of 256 threads (8 waves)
// =====================================================================
__global__ __launch_bounds__(256, 1)
void perf_ctx(const float* __restrict__ qs, const float* __restrict__ proj,
              const float* __restrict__ buf1, float* __restrict__ out) {
  __shared__ __align__(16) f16  pj[Mc * DP];   // proj [m][d]
  __shared__ __align__(16) f16  qt[QC * DP];   // scaled Q chunk [q][d]
  __shared__ __align__(16) f16  b1[EP * CP];   // buf1 transposed [e][m]
  __shared__ float hn[QC];
  __shared__ float dn[QC];

  const int tid   = threadIdx.x;
  const int head  = blockIdx.x >> 5;           // L/QC = 32 chunks
  const int chunk = blockIdx.x & 31;
  const int qbase = chunk * QC;

  // --- stage proj ---
  {
    const int m = tid;
    const float* pr = proj + m * Dc;
    f16* dst = pj + m * DP;
#pragma unroll
    for (int i = 0; i < 16; ++i) {
      f32x4 x = ((const f32x4*)pr)[i];
      h4 y = {(f16)x.x, (f16)x.y, (f16)x.z, (f16)x.w};
      ((h4*)dst)[i] = y;
    }
  }
  // --- stage Q rows (threads 0..127) ---
  if (tid < QC) {
    const int q = tid;
    const float* qr = qs + ((size_t)head * Lc + qbase + q) * Dc;
    f16* dst = qt + q * DP;
    float ss = 0.f;
#pragma unroll
    for (int i = 0; i < 16; ++i) {
      f32x4 x = ((const f32x4*)qr)[i];
      x.x *= NM; x.y *= NM; x.z *= NM; x.w *= NM;
      ss += x.x * x.x + x.y * x.y + x.z * x.z + x.w * x.w;
      h4 y = {(f16)x.x, (f16)x.y, (f16)x.z, (f16)x.w};
      ((h4*)dst)[i] = y;
    }
    hn[q] = 0.5f * ss + LN16;
  }
  // --- stage buf1 transposed (rows 65..79 are zeros from init) ---
  {
    const int m = tid;
    const float* br = buf1 + ((size_t)head * Mc + m) * EP;
#pragma unroll 8
    for (int e = 0; e < EP; ++e) b1[e * CP + m] = (f16)br[e];
  }
  __syncthreads();

  const int lane = tid & 31, w = tid >> 5;
  const int l15 = lane & 15, hi = lane >> 4;
  const int q0 = w * 16;

  // resident B-fragments of q^T (col-major view of [d x q])
  v16h bq[2];
  bq[0] = ldsfrag(qt, q0 + l15, DP, 0, hi);
  bq[1] = ldsfrag(qt, q0 + l15, DP, 32, hi);
  const float hq = hn[q0 + l15];   // lane = q-col: same for all 8 rows

  v8f d2[5];
#pragma unroll
  for (int et = 0; et < 5; ++et) { v8f z = {}; d2[et] = z; }

  for (int t = 0; t < Mc / 32; ++t) {
    const int m0 = t * 32;
    // WMMA1 (transposed): xpT tiles [m x q]
    v8f d1[2];
#pragma unroll
    for (int mi = 0; mi < 2; ++mi) {
      v16h a0 = ldsfrag(pj, m0 + mi * 16 + l15, DP, 0, hi);
      v16h a1 = ldsfrag(pj, m0 + mi * 16 + l15, DP, 32, hi);
      v8f z = {};
      z = wmma(a0, bq[0], z);
      z = wmma(a1, bq[1], z);
      d1[mi] = z;
    }
    // phi -> A2 fragment (rows = q, halves = m)
    v16h a2;
#pragma unroll
    for (int v = 0; v < 8; ++v) {
      a2[v]     = (f16)__expf(d1[0][v] - hq);
      a2[8 + v] = (f16)__expf(d1[1][v] - hq);
    }
    // WMMA2: ctx tiles [q x e] over m
#pragma unroll
    for (int et = 0; et < 5; ++et) {
      v16h b2 = ldsfrag(b1, et * 16 + l15, CP, m0, hi);
      d2[et] = wmma(a2, b2, d2[et]);
    }
  }

  // broadcast denominator column (e == 64 -> lanes with l15 == 0)
  if (l15 == 0) {
#pragma unroll
    for (int v = 0; v < 8; ++v) dn[q0 + hi * 8 + v] = d2[4][v];
  }
  __syncthreads();

  float rd[8];
#pragma unroll
  for (int v = 0; v < 8; ++v)
    rd[v] = 1.f / fmaxf(dn[q0 + hi * 8 + v], EPSV);

  float* orow = out + ((size_t)head * Lc + qbase) * Dc;
#pragma unroll
  for (int et = 0; et < 4; ++et) {
    const int e = et * 16 + l15;
#pragma unroll
    for (int v = 0; v < 8; ++v) {
      const int q = q0 + hi * 8 + v;
      orow[(size_t)q * Dc + e] = d2[et][v] * rd[v];
    }
  }
}

// =====================================================================
__global__ void perf_zero(float* p, int n) {
  int i = blockIdx.x * blockDim.x + threadIdx.x;
  if (i < n) p[i] = 0.f;
}

extern "C" void kernel_launch(void* const* d_in, const int* in_sizes, int n_in,
                              void* d_out, int out_size, void* d_ws, size_t ws_size,
                              hipStream_t stream) {
  const float* qs   = (const float*)d_in[0];
  const float* ks   = (const float*)d_in[1];
  const float* vs   = (const float*)d_in[2];
  const float* mask = (const float*)d_in[3];
  const float* proj = (const float*)d_in[4];
  float* out  = (float*)d_out;
  float* buf1 = (float*)d_ws;                  // [BH][256][80] f32 = 5.0 MiB

  const int nbuf = BH * Mc * EP;
  perf_zero<<<(nbuf + 255) / 256, 256, 0, stream>>>(buf1, nbuf);
  perf_bufk<<<BH * (Lc / SC), 256, 0, stream>>>(ks, vs, mask, proj, buf1);
  perf_ctx <<<BH * (Lc / QC), 256, 0, stream>>>(qs, proj, buf1, out);
}
